// Attention_69544110457546
// MI455X (gfx1250) — matrix operand order repaired and checked by reading, not compiled, not run on hardware
//
#include <hip/hip_runtime.h>

// Problem constants (from reference): TQ=2, TK=5, B=16384, D=512
#define TQ 2
#define TK 5
#define BATCH 16384
#define DIM 512
#define ROW (16 * DIM)   // one 16-batch rowset tile [16][512]

typedef __attribute__((ext_vector_type(16))) __bf16 v16bf;
typedef __attribute__((ext_vector_type(8)))  __bf16 v8bf;
typedef __attribute__((ext_vector_type(8)))  float  v8f;
typedef unsigned short u16;
typedef unsigned int   u32;

// f32 -> bf16 via native fptrunc (RNE); pairs lower to v_cvt_pk_bf16_f32
__device__ __forceinline__ u16 f2bf(float f) {
  union { __bf16 h; u16 u; } c; c.h = (__bf16)f; return c.u;
}
__device__ __forceinline__ u32 pack_bf16(float a, float b) {
  union { __bf16 h[2]; u32 u; } c;
  c.h[0] = (__bf16)a; c.h[1] = (__bf16)b;
  return c.u;
}
// unpack two bf16 from a packed u32 (1 VALU op each)
__device__ __forceinline__ float bflo(u32 p) { union { u32 u; float f; } c; c.u = p << 16;          return c.f; }
__device__ __forceinline__ float bfhi(u32 p) { union { u32 u; float f; } c; c.u = p & 0xFFFF0000u; return c.f; }

// A-matrix fragment (16x32 bf16): lane L<16: M=L, K=0..15;
// lane L>=16: M=L-16, K=16..31. Two contiguous 16B loads per lane.
__device__ __forceinline__ v16bf load_a_frag(const u16* A /*[16][512]*/, int lane, int k0) {
  const int m  = lane & 15;
  const int kb = k0 + ((lane & 16) ? 16 : 0);
  v8bf lo = *(const v8bf*)(A + m * DIM + kb);
  v8bf hi = *(const v8bf*)(A + m * DIM + kb + 8);
  return __builtin_shufflevector(lo, hi, 0,1,2,3,4,5,6,7,8,9,10,11,12,13,14,15);
}

// B-matrix fragment (32x16 bf16) from transposed weights WT[n][k]:
// lane L<16: N=L, K=0..15; lane L>=16: N=L-16, K=16..31 — one contiguous 32B load.
__device__ __forceinline__ v16bf load_b_frag(const u16* WT /*[512][512]*/, int lane, int n0, int k0) {
  const int n  = n0 + (lane & 15);
  const int kb = k0 + ((lane & 16) ? 16 : 0);
  return *(const v16bf*)(WT + (size_t)n * DIM + kb);
}

__device__ __forceinline__ v8f wmma_bf16(v16bf a, v16bf b, v8f c) {
  return __builtin_amdgcn_wmma_f32_16x16x32_bf16(false, a, false, b, (short)0, c, false, false);
}

// ---------------------------------------------------------------------------
// Prologue 1: convert/transpose weights into workspace (bf16).
// ws layout (each 512*512 u16 = 512 KB):
//   [0] WqT    [1] WkT    [2] WdTopT    [3] WvdT (filled by wvd_gemm)
//   [4] WdBotT (temp)     [5] WvR (temp, straight copy of Wv)
// ---------------------------------------------------------------------------
__global__ void weight_prep(const float* __restrict__ Wq, const float* __restrict__ Wk,
                            const float* __restrict__ Wv, const float* __restrict__ Wd,
                            u16* __restrict__ ws) {
  u16* WqT  = ws;
  u16* WkT  = ws + 1 * DIM * DIM;
  u16* WdTT = ws + 2 * DIM * DIM;
  u16* WdBT = ws + 4 * DIM * DIM;
  u16* WvR  = ws + 5 * DIM * DIM;
  const int idx = blockIdx.x * blockDim.x + threadIdx.x;
  if (idx >= DIM * DIM) return;
  const int n = idx >> 9;
  const int k = idx & (DIM - 1);
  WqT[idx]  = f2bf(Wq[k * DIM + n]);
  WkT[idx]  = f2bf(Wk[k * DIM + n]);
  WdTT[idx] = f2bf(Wd[k * DIM + n]);            // Wdown rows [0,512), transposed
  WdBT[idx] = f2bf(Wd[(DIM + k) * DIM + n]);    // Wdown rows [512,1024), transposed
  WvR[idx]  = f2bf(Wv[idx]);                    // row-major copy
}

// ---------------------------------------------------------------------------
// Prologue 2: WvdT[n][k] = (Wv @ Wdown_bot)^T via WMMA. 1024 16x16 tiles,
// one wave each (128 blocks x 8 waves).
// ---------------------------------------------------------------------------
__global__ void __launch_bounds__(256) wvd_gemm(u16* __restrict__ ws) {
  const u16* WvR  = ws + 5 * DIM * DIM;  // A: [k][j]
  const u16* WdBT = ws + 4 * DIM * DIM;  // B (transposed): [n][j]
  u16* WvdT       = ws + 3 * DIM * DIM;  // out: [n][k]
  const int lane = threadIdx.x & 31;
  const int wave = threadIdx.x >> 5;
  const int tile = blockIdx.x * 8 + wave;     // 0..1023
  const int m0 = (tile & 31) * 16;            // k-dim of G = Wv@WdBot
  const int n0 = (tile >> 5) * 16;
  v8f acc = {};
  for (int k0 = 0; k0 < DIM; k0 += 32) {
    v16bf a = load_a_frag(WvR + m0 * DIM, lane, k0);
    v16bf b = load_b_frag(WdBT, lane, n0, k0);
    acc = wmma_bf16(a, b, acc);
  }
  // D row for this lane is contiguous in k -> one 16B packed store
  const int n  = n0 + (lane & 15);
  const int mb = m0 + ((lane & 16) ? 8 : 0);
  uint4 o;
  o.x = pack_bf16(acc[0], acc[1]); o.y = pack_bf16(acc[2], acc[3]);
  o.z = pack_bf16(acc[4], acc[5]); o.w = pack_bf16(acc[6], acc[7]);
  *(uint4*)(WvdT + (size_t)n * DIM + mb) = o;
}

// ---------------------------------------------------------------------------
// Main fused kernel: one block = 16 batches, 8 wave32s each owning 64 features.
// qk scores are harvested directly from the Q/K WMMA accumulators (no Q/K
// materialization): lane-local products -> shfl_xor butterfly over the 16
// n-lanes -> ds_add_f32 across the 8 feature-slice waves.
// ---------------------------------------------------------------------------
__global__ void __launch_bounds__(256) fused_attention(
    const float* __restrict__ h, const float* __restrict__ enc,
    const u16* __restrict__ ws, float* __restrict__ out) {
  extern __shared__ __align__(16) char smem_raw[];
  u16* Hs    = (u16*)smem_raw;            // [2][16][512] bf16   32 KB
  u16* Es    = Hs + TQ * ROW;             // [5][16][512]        80 KB
  u16* Ebars = Es + TK * ROW;             // [2][16][512]        32 KB
  float* qk  = (float*)(Ebars + TQ * ROW);// [16][2][5]
  float* att = qk + 16 * TQ * TK;         // [16][2][5]

  const u16* WqT  = ws;
  const u16* WkT  = ws + 1 * DIM * DIM;
  const u16* WdTT = ws + 2 * DIM * DIM;
  const u16* WvdT = ws + 3 * DIM * DIM;

  const int tid  = threadIdx.x;
  const int lane = tid & 31;
  const int wave = tid >> 5;
  const int b0   = blockIdx.x * 16;

  // ---- Stage 1: stream activations (contiguous 32KB chunks), cvt to bf16 in LDS
  if (tid < 16 * TQ * TK) qk[tid] = 0.f;   // accumulation target for stage 2
#pragma unroll
  for (int t = 0; t < TQ; ++t) {
    const float4* src = (const float4*)(h + (size_t)t * BATCH * DIM + (size_t)b0 * DIM);
    u16* dst = Hs + t * ROW;
    if (tid == 0 && b0 + 16 < BATCH)
      __builtin_prefetch(h + (size_t)t * BATCH * DIM + (size_t)(b0 + 16) * DIM, 0, 1);
    for (int i = tid; i < ROW / 4; i += 256) {
      float4 v = src[i];
      ((u32*)dst)[2 * i]     = pack_bf16(v.x, v.y);
      ((u32*)dst)[2 * i + 1] = pack_bf16(v.z, v.w);
    }
  }
#pragma unroll
  for (int s = 0; s < TK; ++s) {
    const float4* src = (const float4*)(enc + (size_t)s * BATCH * DIM + (size_t)b0 * DIM);
    u16* dst = Es + s * ROW;
    if (tid == 0 && b0 + 16 < BATCH)
      __builtin_prefetch(enc + (size_t)s * BATCH * DIM + (size_t)(b0 + 16) * DIM, 0, 1);
    for (int i = tid; i < ROW / 4; i += 256) {
      float4 v = src[i];
      ((u32*)dst)[2 * i]     = pack_bf16(v.x, v.y);
      ((u32*)dst)[2 * i + 1] = pack_bf16(v.z, v.w);
    }
  }
  __syncthreads();

  // ---- Stage 2: Q = H@Wq, K = E@Wk (never materialized); qk dots from accumulators.
  float dp[TQ][TK][8];
#pragma unroll
  for (int t = 0; t < TQ; ++t)
#pragma unroll
    for (int s = 0; s < TK; ++s)
#pragma unroll
      for (int j = 0; j < 8; ++j) dp[t][s][j] = 0.f;

  for (int nt = 0; nt < 4; ++nt) {
    const int n0 = wave * 64 + nt * 16;
    const v8f zero = {};
    v8f aq[TQ], ak[TK];
#pragma unroll
    for (int t = 0; t < TQ; ++t) aq[t] = zero;
#pragma unroll
    for (int s = 0; s < TK; ++s) ak[s] = zero;
    for (int k0 = 0; k0 < DIM; k0 += 32) {
      const v16bf bq = load_b_frag(WqT, lane, n0, k0);
      const v16bf bk = load_b_frag(WkT, lane, n0, k0);
#pragma unroll
      for (int t = 0; t < TQ; ++t)
        aq[t] = wmma_bf16(load_a_frag(Hs + t * ROW, lane, k0), bq, aq[t]);
#pragma unroll
      for (int s = 0; s < TK; ++s)
        ak[s] = wmma_bf16(load_a_frag(Es + s * ROW, lane, k0), bk, ak[s]);
    }
    // accumulator element (j, lane) = (batch i = mb+j, feature n): elementwise dot part
#pragma unroll
    for (int t = 0; t < TQ; ++t)
#pragma unroll
      for (int s = 0; s < TK; ++s)
#pragma unroll
        for (int j = 0; j < 8; ++j)
          dp[t][s][j] += aq[t][j] * ak[s][j];
  }
  // butterfly-reduce over the 16 n-lanes of each half-wave
#pragma unroll
  for (int t = 0; t < TQ; ++t)
#pragma unroll
    for (int s = 0; s < TK; ++s)
#pragma unroll
      for (int j = 0; j < 8; ++j) {
        float v = dp[t][s][j];
        v += __shfl_xor(v, 1);
        v += __shfl_xor(v, 2);
        v += __shfl_xor(v, 4);
        v += __shfl_xor(v, 8);
        dp[t][s][j] = v;
      }
  if ((lane & 15) == 0) {
    const int mb = (lane & 16) ? 8 : 0;
#pragma unroll
    for (int t = 0; t < TQ; ++t)
#pragma unroll
      for (int s = 0; s < TK; ++s)
#pragma unroll
        for (int j = 0; j < 8; ++j)
          atomicAdd(&qk[(mb + j) * (TQ * TK) + t * TK + s], dp[t][s][j]);
  }
  __syncthreads();

  // ---- Stage 3: softmax over s (one thread per (batch, t))
  if (tid < 16 * TQ) {
    const float* row = qk + tid * TK;
    float m = row[0];
#pragma unroll
    for (int s = 1; s < TK; ++s) m = fmaxf(m, row[s]);
    float e[TK], sum = 0.f;
#pragma unroll
    for (int s = 0; s < TK; ++s) { e[s] = __expf(row[s] - m); sum += e[s]; }
    const float inv = 1.f / sum;
    float* arow = att + tid * TK;
#pragma unroll
    for (int s = 0; s < TK; ++s) arow[s] = e[s] * inv;
  }
  __syncthreads();

  // ---- Stage 4: Ebar[t] = sum_s att[t,s] * E[s]   (8 elements per thread-iter)
  for (int idx = tid; idx < (TQ * ROW) / 8; idx += 256) {
    const int d8 = idx & (DIM / 8 - 1);       // 0..63
    const int i  = (idx >> 6) & 15;
    const int t  = idx >> 10;
    const float* arow = att + (i * TQ + t) * TK;
    float a0 = 0, a1 = 0, a2 = 0, a3 = 0, a4 = 0, a5 = 0, a6 = 0, a7 = 0;
#pragma unroll
    for (int s = 0; s < TK; ++s) {
      const uint4 e = *(const uint4*)(Es + s * ROW + i * DIM + d8 * 8);
      const float w = arow[s];
      a0 += w * bflo(e.x); a1 += w * bfhi(e.x);
      a2 += w * bflo(e.y); a3 += w * bfhi(e.y);
      a4 += w * bflo(e.z); a5 += w * bfhi(e.z);
      a6 += w * bflo(e.w); a7 += w * bfhi(e.w);
    }
    uint4 o;
    o.x = pack_bf16(a0, a1); o.y = pack_bf16(a2, a3);
    o.z = pack_bf16(a4, a5); o.w = pack_bf16(a6, a7);
    *(uint4*)(Ebars + t * ROW + i * DIM + d8 * 8) = o;
  }
  __syncthreads();

  // ---- Stage 5: out[t] = H[t]@WdownTop + Ebar[t]@(Wv@WdownBot), f32 stores
  for (int nt = 0; nt < 4; ++nt) {
    const int n0 = wave * 64 + nt * 16;
    v8f o0 = {}, o1 = {};
    for (int k0 = 0; k0 < DIM; k0 += 32) {
      v16bf b1 = load_b_frag(WdTT, lane, n0, k0);
      v16bf b2 = load_b_frag(WvdT, lane, n0, k0);
      o0 = wmma_bf16(load_a_frag(Hs + 0 * ROW,    lane, k0), b1, o0);
      o0 = wmma_bf16(load_a_frag(Ebars + 0 * ROW, lane, k0), b2, o0);
      o1 = wmma_bf16(load_a_frag(Hs + 1 * ROW,    lane, k0), b1, o1);
      o1 = wmma_bf16(load_a_frag(Ebars + 1 * ROW, lane, k0), b2, o1);
    }
    const int n  = n0 + (lane & 15);
    const int mb = (lane & 16) ? 8 : 0;
#pragma unroll
    for (int j = 0; j < 8; ++j) {
      out[(size_t)0 * BATCH * DIM + (size_t)(b0 + mb + j) * DIM + n] = o0[j];
      out[(size_t)1 * BATCH * DIM + (size_t)(b0 + mb + j) * DIM + n] = o1[j];
    }
  }
}

// ---------------------------------------------------------------------------
extern "C" void kernel_launch(void* const* d_in, const int* in_sizes, int n_in,
                              void* d_out, int out_size, void* d_ws, size_t ws_size,
                              hipStream_t stream) {
  (void)in_sizes; (void)n_in; (void)out_size; (void)ws_size;
  const float* h   = (const float*)d_in[0];
  const float* enc = (const float*)d_in[1];
  const float* Wq  = (const float*)d_in[2];
  const float* Wk  = (const float*)d_in[3];
  const float* Wv  = (const float*)d_in[4];
  const float* Wd  = (const float*)d_in[5];
  float* out = (float*)d_out;
  u16* ws = (u16*)d_ws;  // needs 6 * 512KB = 3 MB

  weight_prep<<<(DIM * DIM) / 256, 256, 0, stream>>>(Wq, Wk, Wv, Wd, ws);
  wvd_gemm<<<128, 256, 0, stream>>>(ws);

  const int smem_bytes = (TQ + TK + TQ) * ROW * (int)sizeof(u16)
                       + 2 * 16 * TQ * TK * (int)sizeof(float);  // 148,736 B -> 2 WGs/WGP
  fused_attention<<<BATCH / 16, 256, smem_bytes, stream>>>(h, enc, ws, out);
}
